// AgentAwareAttention_32573031972945
// MI455X (gfx1250) — matrix operand order repaired and checked by reading, not compiled, not run on hardware
//
#include <hip/hip_runtime.h>
#include <hip/hip_bf16.h>

typedef __attribute__((ext_vector_type(16))) __bf16 v16bf;
typedef __attribute__((ext_vector_type(8)))  float  v8f;

namespace {

constexpr int    BATCH  = 4;
constexpr int    NTOK   = 2009;
constexpr int    NPAD   = 2048;          // padded token count (multiple of 64)
constexpr int    DMODEL = 1024;
constexpr int    NHEADS = 4;
constexpr int    BH     = BATCH * NHEADS;
constexpr float  SCALE  = 0.0625f;       // 256^-0.5
constexpr size_t OUT_ELEMS = (size_t)BATCH * NTOK * DMODEL;

// 32 payload dwords + 4 pad dwords: 144 B rows -> 16-byte aligned rows, and
// 36*m mod 64 is distinct for m=0..15 so fragment reads are bank-conflict-free.
#define LDS_STRIDE 36

// ---- CDNA5 async global->LDS copy (ASYNCcnt-tracked, no VGPR round-trip) ---
__device__ inline void async_copy_b128(unsigned int* ldsDst, const unsigned int* gsrc) {
  // Low 32 bits of a flat shared pointer are the wave-relative LDS byte offset.
  unsigned int ldsAddr = (unsigned int)(unsigned long long)(void*)ldsDst;
  unsigned long long ga = (unsigned long long)gsrc;
  asm volatile("global_load_async_to_lds_b128 %0, %1, off"
               :: "v"(ldsAddr), "v"(ga) : "memory");
}
__device__ inline void wait_async0() {
  asm volatile("s_wait_asynccnt 0x0" ::: "memory");
}

// ---- WMMA bf16 fragment loaders (per CDNA5 ISA 7.12.2 layouts) -------------
// A: 16x32 bf16. lane&15 = M row, lane>>4 selects K half. Dwords are two
// contiguous 16B groups -> two ds_load_b128.
__device__ inline v16bf load_a_frag(const unsigned int* lds, int rowBase, int lane, int kdw) {
  union { uint4 q[2]; v16bf v; } f;
  const int m  = lane & 15;
  const int hi = lane >> 4;
  const unsigned int* p = &lds[(rowBase + m) * LDS_STRIDE + (hi << 2) + kdw];
  f.q[0] = *(const uint4*)p;
  f.q[1] = *(const uint4*)(p + 8);
  return f.v;
}

// B: 32x16 bf16 stored row-major N x K in LDS; lanes 0-15 K=0..7,16..23, 16-31 K=8..15,24..31.
// two 16B groups -> two ds_load_b128.
__device__ inline v16bf load_b_frag(const unsigned int* lds, int colBase, int lane, int kdw) {
  union { uint4 q[2]; v16bf v; } f;
  const int n  = lane & 15;
  const int hi = lane >> 4;
  const unsigned int* p = &lds[(colBase + n) * LDS_STRIDE + (hi << 2) + kdw];
  f.q[0] = *(const uint4*)p;
  f.q[1] = *(const uint4*)(p + 8);
  return f.v;
}

#define WMMA_BF16(A, B, C) \
  __builtin_amdgcn_wmma_f32_16x16x32_bf16(false, (A), false, (B), (short)0, (C), false, false)

// Fill one 64-row x 32-dword LDS tile from global via async b128 copies.
// 512 chunks per tile, 128 threads -> 4 chunks each. Caller must
// wait_async0() + __syncthreads() before consuming.
#define FILL_TILE_ASYNC(ldsArr, gptr, rowStrideDw, rowOff, colOffDw)                     \
  _Pragma("unroll")                                                                      \
  for (int i = 0; i < 4; ++i) {                                                          \
    int idx = t + i * 128;                                                               \
    int rr = idx >> 3, c4 = idx & 7;                                                     \
    async_copy_b128(&ldsArr[rr * LDS_STRIDE + c4 * 4],                                   \
        &(gptr)[(size_t)((rowOff) + rr) * (rowStrideDw) + (colOffDw) + c4 * 4]);         \
  }

// ---------------------------------------------------------------------------
// Conversion kernels
// ---------------------------------------------------------------------------
__global__ __launch_bounds__(256)
void convx_kernel(const float* __restrict__ x, __bf16* __restrict__ xb) {
  size_t t4 = (size_t)blockIdx.x * 256 + threadIdx.x;      // over 8192*1024/4
  if (t4 >= (size_t)BATCH * NPAD * DMODEL / 4) return;
  int c4   = (int)(t4 & 255);                              // 256 float4 per row
  int rowp = (int)(t4 >> 8);
  int b = rowp >> 11;
  int n = rowp & (NPAD - 1);
  float4 v = make_float4(0.f, 0.f, 0.f, 0.f);
  if (n < NTOK) v = *(const float4*)&x[((size_t)b * NTOK + n) * DMODEL + c4 * 4];
  union { __bf16 h[4]; uint2 u; } pk;
  pk.h[0] = (__bf16)v.x; pk.h[1] = (__bf16)v.y;
  pk.h[2] = (__bf16)v.z; pk.h[3] = (__bf16)v.w;
  *(uint2*)&xb[t4 * 4] = pk.u;
}

// src [K][C] f32 -> dst [C][K] bf16 (transpose + convert; tiny, scalar ok)
__global__ __launch_bounds__(256)
void convw_kernel(const float* __restrict__ src, __bf16* __restrict__ dst, int K, int C) {
  int tid = blockIdx.x * 256 + threadIdx.x;
  if (tid >= K * C) return;
  int k = tid % K;
  int c = tid / K;
  dst[tid] = (__bf16)src[(size_t)k * C + c];
}

// ---------------------------------------------------------------------------
// Projection GEMM: Xb[8192,1024]bf16 @ Wt[ncols,1024]^T, scatter to head layouts
// mode 0: cols 0-255 -> Q[bh][n][64], 256-511 -> K[bh][n][64], 512-767 -> Vt[bh][64][n]
// mode 1: cols 0-255 -> QS,           256-511 -> KS
// ---------------------------------------------------------------------------
__global__ __launch_bounds__(128)
void proj_kernel(const unsigned int* __restrict__ Xb,  // dwords, row stride 512
                 const unsigned int* __restrict__ Wt,  // dwords, row stride 512
                 __bf16* __restrict__ d0, __bf16* __restrict__ d1,
                 __bf16* __restrict__ d2, int mode) {
  __shared__ __align__(16) unsigned int lA[64 * LDS_STRIDE];
  __shared__ __align__(16) unsigned int lB[64 * LDS_STRIDE];
  const int t = threadIdx.x, wave = t >> 5, lane = t & 31;
  const int rowBase = blockIdx.x * 64, colBase = blockIdx.y * 64;

  v8f acc[4];
#pragma unroll
  for (int c = 0; c < 4; ++c) acc[c] = {};

  for (int kc = 0; kc < 16; ++kc) {                 // K = 1024 in chunks of 64
    FILL_TILE_ASYNC(lA, Xb, 512, rowBase, kc * 32);
    FILL_TILE_ASYNC(lB, Wt, 512, colBase, kc * 32);
    wait_async0();
    __syncthreads();
#pragma unroll
    for (int kd = 0; kd < 32; kd += 16) {
      v16bf a = load_a_frag(lA, wave * 16, lane, kd);
#pragma unroll
      for (int c = 0; c < 4; ++c) {
        v16bf b = load_b_frag(lB, c * 16, lane, kd);
        acc[c] = WMMA_BF16(a, b, acc[c]);
      }
    }
    __syncthreads();
  }

  const int hi = lane >> 4, nn = lane & 15;
#pragma unroll
  for (int c = 0; c < 4; ++c)
#pragma unroll
    for (int r = 0; r < 8; ++r) {
      int row = rowBase + wave * 16 + hi * 8 + r;
      int col = colBase + c * 16 + nn;
      __bf16 v = (__bf16)acc[c][r];
      int b = row >> 11, n = row & (NPAD - 1);
      if (mode == 0) {
        if (col < 256)
          d0[(((size_t)b * 4 + (col >> 6)) * NPAD + n) * 64 + (col & 63)] = v;
        else if (col < 512)
          d1[(((size_t)b * 4 + ((col - 256) >> 6)) * NPAD + n) * 64 + ((col - 256) & 63)] = v;
        else
          d2[(((size_t)b * 4 + ((col - 512) >> 6)) * 64 + ((col - 512) & 63)) * NPAD + n] = v;
      } else {
        if (col < 256)
          d0[(((size_t)b * 4 + (col >> 6)) * NPAD + n) * 64 + (col & 63)] = v;
        else
          d1[(((size_t)b * 4 + ((col - 256) >> 6)) * NPAD + n) * 64 + ((col - 256) & 63)] = v;
      }
    }
}

// ---------------------------------------------------------------------------
// dots = Q @ K^T * SCALE  -> attn buffer (f32, [bh][2009][2009])
// ---------------------------------------------------------------------------
__global__ __launch_bounds__(128)
void dots_kernel(const unsigned int* __restrict__ Qb,   // [bh][NPAD][32] dwords
                 const unsigned int* __restrict__ Kb,
                 float* __restrict__ attn) {
  __shared__ __align__(16) unsigned int lA[64 * LDS_STRIDE];
  __shared__ __align__(16) unsigned int lB[64 * LDS_STRIDE];
  const int t = threadIdx.x, wave = t >> 5, lane = t & 31;
  const int bh = blockIdx.z;
  const int rowBase = blockIdx.x * 64, colBase = blockIdx.y * 64;
  const unsigned int* Qp = Qb + (size_t)bh * NPAD * 32;
  const unsigned int* Kp = Kb + (size_t)bh * NPAD * 32;

  FILL_TILE_ASYNC(lA, Qp, 32, rowBase, 0);
  FILL_TILE_ASYNC(lB, Kp, 32, colBase, 0);
  wait_async0();
  __syncthreads();

  v8f acc[4];
#pragma unroll
  for (int c = 0; c < 4; ++c) acc[c] = {};
#pragma unroll
  for (int kd = 0; kd < 32; kd += 16) {             // per-head dim K = 64
    v16bf a = load_a_frag(lA, wave * 16, lane, kd);
#pragma unroll
    for (int c = 0; c < 4; ++c) {
      v16bf b = load_b_frag(lB, c * 16, lane, kd);
      acc[c] = WMMA_BF16(a, b, acc[c]);
    }
  }

  const int hi = lane >> 4, nn = lane & 15;
#pragma unroll
  for (int c = 0; c < 4; ++c)
#pragma unroll
    for (int r = 0; r < 8; ++r) {
      int row = rowBase + wave * 16 + hi * 8 + r;
      int col = colBase + c * 16 + nn;
      if (row < NTOK && col < NTOK)
        attn[((size_t)bh * NTOK + row) * NTOK + col] = acc[c][r] * SCALE;
    }
}

// ---------------------------------------------------------------------------
// Patch 10x10 agent diagonal blocks with q_self . k_self * SCALE
// ---------------------------------------------------------------------------
__global__ __launch_bounds__(256)
void patch_kernel(const __bf16* __restrict__ QS, const __bf16* __restrict__ KS,
                  float* __restrict__ attn) {
  int idx = blockIdx.x * 256 + threadIdx.x;
  if (idx >= BH * 200 * 100) return;                 // 320000 masked elements
  int j = idx % 10; int tmp = idx / 10;
  int i = tmp % 10; tmp /= 10;
  int a = tmp % 200; int bh = tmp / 200;
  int row = a * 10 + i, col = a * 10 + j;
  const uint4* q4 = (const uint4*)(QS + ((size_t)bh * NPAD + row) * 64);
  const uint4* k4 = (const uint4*)(KS + ((size_t)bh * NPAD + col) * 64);
  float s = 0.0f;
#pragma unroll
  for (int v = 0; v < 8; ++v) {
    union { uint4 q; __bf16 h[8]; } aa, bb;
    aa.q = q4[v]; bb.q = k4[v];
#pragma unroll
    for (int d = 0; d < 8; ++d) s += (float)aa.h[d] * (float)bb.h[d];
  }
  attn[((size_t)bh * NTOK + row) * NTOK + col] = s * SCALE;
}

// ---------------------------------------------------------------------------
// Row softmax over 2009 columns, in place (one workgroup per row)
// ---------------------------------------------------------------------------
__global__ __launch_bounds__(256)
void softmax_kernel(float* __restrict__ attn) {
  __shared__ float red[256];
  float* row = attn + (size_t)blockIdx.x * NTOK;
  const int t = threadIdx.x;

  float m = -3.0e38f;
  for (int j = t; j < NTOK; j += 256) m = fmaxf(m, row[j]);
  red[t] = m; __syncthreads();
  for (int s = 128; s > 0; s >>= 1) { if (t < s) red[t] = fmaxf(red[t], red[t + s]); __syncthreads(); }
  const float M = red[0];
  __syncthreads();

  float sum = 0.0f;
  for (int j = t; j < NTOK; j += 256) { float e = __expf(row[j] - M); row[j] = e; sum += e; }
  red[t] = sum; __syncthreads();
  for (int s = 128; s > 0; s >>= 1) { if (t < s) red[t] += red[t + s]; __syncthreads(); }
  const float inv = 1.0f / red[0];
  __syncthreads();

  for (int j = t; j < NTOK; j += 256) row[j] *= inv;
}

// ---------------------------------------------------------------------------
// attn @ V : per (bh), [n,m] x [m,64] -> HO[(b*NPAD+n)][h*64+d] bf16
// attn read as f32, converted to bf16 while packing A fragments;
// V^T tile streamed via async LDS copies.
// ---------------------------------------------------------------------------
__global__ __launch_bounds__(128)
void attnv_kernel(const float* __restrict__ attn,
                  const unsigned int* __restrict__ Vt,   // [bh][64][1024] dwords
                  __bf16* __restrict__ HO) {
  __shared__ __align__(16) unsigned int lA[64 * LDS_STRIDE];
  __shared__ __align__(16) unsigned int lB[64 * LDS_STRIDE];
  const int t = threadIdx.x, wave = t >> 5, lane = t & 31;
  const int bh = blockIdx.z;
  const int b = bh >> 2, h = bh & 3;
  const int rowBase = blockIdx.x * 64;
  const unsigned int* Vp = Vt + (size_t)bh * 64 * 1024;

  v8f acc[4];
#pragma unroll
  for (int c = 0; c < 4; ++c) acc[c] = {};

  for (int kc = 0; kc < 32; ++kc) {                  // m = 2048 in chunks of 64
    FILL_TILE_ASYNC(lB, Vp, 1024, 0, kc * 32);
    // A tile: f32 attn -> packed bf16 pairs (rows guarded, cols >= NTOK -> 0)
#pragma unroll
    for (int i = 0; i < 16; ++i) {
      int idx = t + i * 128;
      int rr = idx >> 5, cc = idx & 31;
      int r  = rowBase + rr;
      int m0 = kc * 64 + cc * 2;
      float f0 = (r < NTOK && m0     < NTOK) ? attn[((size_t)bh * NTOK + r) * NTOK + m0]     : 0.0f;
      float f1 = (r < NTOK && m0 + 1 < NTOK) ? attn[((size_t)bh * NTOK + r) * NTOK + m0 + 1] : 0.0f;
      union { __bf16 h2[2]; unsigned int u; } pk;
      pk.h2[0] = (__bf16)f0; pk.h2[1] = (__bf16)f1;
      lA[rr * LDS_STRIDE + cc] = pk.u;
    }
    wait_async0();
    __syncthreads();
#pragma unroll
    for (int kd = 0; kd < 32; kd += 16) {
      v16bf a = load_a_frag(lA, wave * 16, lane, kd);
#pragma unroll
      for (int c = 0; c < 4; ++c) {
        v16bf bfr = load_b_frag(lB, c * 16, lane, kd);
        acc[c] = WMMA_BF16(a, bfr, acc[c]);
      }
    }
    __syncthreads();
  }

  const int hi = lane >> 4, nn = lane & 15;
#pragma unroll
  for (int c = 0; c < 4; ++c)
#pragma unroll
    for (int r = 0; r < 8; ++r) {
      int row = rowBase + wave * 16 + hi * 8 + r;
      int col = c * 16 + nn;                          // d in 0..63
      HO[((size_t)b * NPAD + row) * 256 + h * 64 + col] = (__bf16)acc[c][r];
    }
}

// ---------------------------------------------------------------------------
// Output projection: HO[8192,256] @ Wout^T + bias -> out (f32)
// ---------------------------------------------------------------------------
__global__ __launch_bounds__(128)
void outproj_kernel(const unsigned int* __restrict__ HO,    // row stride 128 dwords
                    const unsigned int* __restrict__ WoT,   // [1024][128] dwords
                    const float* __restrict__ bias,
                    float* __restrict__ out) {
  __shared__ __align__(16) unsigned int lA[64 * LDS_STRIDE];
  __shared__ __align__(16) unsigned int lB[64 * LDS_STRIDE];
  const int t = threadIdx.x, wave = t >> 5, lane = t & 31;
  const int rowBase = blockIdx.x * 64, colBase = blockIdx.y * 64;

  v8f acc[4];
#pragma unroll
  for (int c = 0; c < 4; ++c) acc[c] = {};

  for (int kc = 0; kc < 4; ++kc) {                   // K = 256 in chunks of 64
    FILL_TILE_ASYNC(lA, HO, 128, rowBase, kc * 32);
    FILL_TILE_ASYNC(lB, WoT, 128, colBase, kc * 32);
    wait_async0();
    __syncthreads();
#pragma unroll
    for (int kd = 0; kd < 32; kd += 16) {
      v16bf a = load_a_frag(lA, wave * 16, lane, kd);
#pragma unroll
      for (int c = 0; c < 4; ++c) {
        v16bf bfr = load_b_frag(lB, c * 16, lane, kd);
        acc[c] = WMMA_BF16(a, bfr, acc[c]);
      }
    }
    __syncthreads();
  }

  const int hi = lane >> 4, nn = lane & 15;
#pragma unroll
  for (int c = 0; c < 4; ++c)
#pragma unroll
    for (int r = 0; r < 8; ++r) {
      int row = rowBase + wave * 16 + hi * 8 + r;
      int col = colBase + c * 16 + nn;
      int b = row >> 11, n = row & (NPAD - 1);
      if (n < NTOK)
        out[((size_t)b * NTOK + n) * DMODEL + col] = acc[c][r] + bias[col];
    }
}

} // anonymous namespace

// ---------------------------------------------------------------------------
extern "C" void kernel_launch(void* const* d_in, const int* in_sizes, int n_in,
                              void* d_out, int out_size, void* d_ws, size_t ws_size,
                              hipStream_t stream) {
  const float* x      = (const float*)d_in[0];   // [4,2009,1024]
  const float* W_qkv  = (const float*)d_in[1];   // [1024,768]
  const float* W_qks  = (const float*)d_in[2];   // [1024,512]
  const float* W_out  = (const float*)d_in[3];   // [256,1024]
  const float* b_out  = (const float*)d_in[4];   // [1024]

  float* out  = (float*)d_out;                   // [4,2009,1024]
  float* attn = (float*)d_out + OUT_ELEMS;       // [4,4,2009,2009]

  // Workspace layout (bytes, all 16B aligned)
  char* ws = (char*)d_ws;
  __bf16* xb    = (__bf16*)(ws);                         // 8192*1024       = 16 MiB
  __bf16* wqkvT = (__bf16*)(ws + 16777216);              // 768*1024
  __bf16* wqksT = (__bf16*)(ws + 18350080);              // 512*1024
  __bf16* woutT = (__bf16*)(ws + 19398656);              // 1024*256
  __bf16* Qb    = (__bf16*)(ws + 19922944);              // 16*2048*64
  __bf16* Kb    = (__bf16*)(ws + 24117248);
  __bf16* Vt    = (__bf16*)(ws + 28311552);              // 16*64*2048
  __bf16* QSb   = (__bf16*)(ws + 32505856);
  __bf16* KSb   = (__bf16*)(ws + 36700160);
  __bf16* HO    = (__bf16*)(ws + 40894464);              // 8192*256  -> ends ~45 MiB

  // 1) convert/pad/transpose inputs to bf16
  convx_kernel<<<(BATCH * NPAD * DMODEL / 4 + 255) / 256, 256, 0, stream>>>(x, xb);
  convw_kernel<<<(1024 * 768) / 256, 256, 0, stream>>>(W_qkv, wqkvT, 1024, 768);
  convw_kernel<<<(1024 * 512) / 256, 256, 0, stream>>>(W_qks, wqksT, 1024, 512);
  convw_kernel<<<(256 * 1024) / 256, 256, 0, stream>>>(W_out, woutT, 256, 1024);

  // 2) projections (WMMA)
  proj_kernel<<<dim3(128, 12, 1), 128, 0, stream>>>(
      (const unsigned int*)xb, (const unsigned int*)wqkvT, Qb, Kb, Vt, 0);
  proj_kernel<<<dim3(128, 8, 1), 128, 0, stream>>>(
      (const unsigned int*)xb, (const unsigned int*)wqksT, QSb, KSb, nullptr, 1);

  // 3) scores = Q K^T * scale (WMMA), then patch agent diagonal blocks
  dots_kernel<<<dim3(32, 32, BH), 128, 0, stream>>>(
      (const unsigned int*)Qb, (const unsigned int*)Kb, attn);
  patch_kernel<<<(BH * 200 * 100 + 255) / 256, 256, 0, stream>>>(QSb, KSb, attn);

  // 4) softmax rows in place
  softmax_kernel<<<BH * NTOK, 256, 0, stream>>>(attn);

  // 5) attn @ V (WMMA), heads concatenated into HO
  attnv_kernel<<<dim3(32, 1, BH), 128, 0, stream>>>(attn, (const unsigned int*)Vt, HO);

  // 6) output projection + bias (WMMA)
  outproj_kernel<<<dim3(128, 16, 1), 128, 0, stream>>>(
      (const unsigned int*)HO, (const unsigned int*)woutT, b_out, out);
}
